// MappingNet_44504451121189
// MI455X (gfx1250) — matrix operand order repaired and checked by reading, not compiled, not run on hardware
//
#include <hip/hip_runtime.h>

#define NEXP   10
#define DIM    1024
#define INDIM  512
#define BATCH  4096
#define KSTEP  32
#define LDA    48            // padded LDS row stride (halfs): 96B, 16B-aligned, reduces bank conflicts
#define BM     128
#define BN     128
#define MAXPAD (BATCH + NEXP * BM)   // 5376: segments padded to 128-row granularity

#if __has_builtin(__builtin_amdgcn_global_load_async_to_lds_b128) && \
    __has_builtin(__builtin_amdgcn_s_wait_asynccnt)
#define USE_ASYNC_LDS 1
#else
#define USE_ASYNC_LDS 0
#endif

#define ASG __attribute__((address_space(1)))
#define ASL __attribute__((address_space(3)))

typedef __attribute__((ext_vector_type(16))) _Float16     v16h;
typedef __attribute__((ext_vector_type(4)))  _Float16     h4;
typedef __attribute__((ext_vector_type(8)))  float        v8f;
typedef __attribute__((ext_vector_type(4)))  float        f32x4;
typedef __attribute__((ext_vector_type(4)))  unsigned int u32x4;
typedef int i32x4 __attribute__((vector_size(16)));   // matches builtin's int4 param type

union Frag { v16h v; u32x4 q[2]; };

__device__ __forceinline__ float gelu_exact(float v) {
    return 0.5f * v * (1.0f + erff(v * 0.70710678118654752f));
}

// A fragment (16x32 f16). ISA layout: lanes 0-15 -> M=lane, halves 0..7 = K 0..7, halves 8..15 = K 16..23;
// lanes 16-31 -> M=lane-16, K offset +8.
__device__ __forceinline__ v16h load_frag_a(const _Float16* As, int mtile, int lane) {
    const _Float16* p = As + (mtile * 16 + (lane & 15)) * LDA + ((lane >> 4) * 8);
    Frag f;
    f.q[0] = *(const u32x4*)p;
    f.q[1] = *(const u32x4*)(p + 16);
    return f.v;
}

// B fragment (32x16 f16), LDS stored [n][k]. ISA layout: lane n=lane%16, halves 0..7 = K 0..7, halves 8..15 = K 16..23; lanes 16-31 K offset +8.
__device__ __forceinline__ v16h load_frag_b(const _Float16* Bs, int ntile, int lane) {
    const _Float16* p = Bs + (ntile * 16 + (lane & 15)) * LDA + ((lane >> 4) * 8);
    Frag f;
    f.q[0] = *(const u32x4*)p;
    f.q[1] = *(const u32x4*)(p + 16);
    return f.v;
}

// Stage one 128x32 A tile (f16 global -> LDS, async when available) and one
// 32x128 B tile (f32 global -> f16 LDS, register-transposed, packed b64 stores).
__device__ __forceinline__ void stage_tiles(const _Float16* __restrict__ Ag,
                                            const float* __restrict__ Bg,
                                            int K, int N, int kbase, int grow, int nbase,
                                            _Float16* As, _Float16* Bs, int tid) {
    // ---- A: thread covers row r = tid/2, 16 halfs at ksel = (tid&1)*16
    {
        const int r    = tid >> 1;
        const int ksel = (tid & 1) * 16;
        _Float16* dst  = As + r * LDA + ksel;
        if (grow >= 0) {
            const _Float16* src = Ag + (size_t)grow * K + kbase + ksel;
#if USE_ASYNC_LDS
            __builtin_amdgcn_global_load_async_to_lds_b128((ASG i32x4*)src,
                                                           (ASL i32x4*)dst, 0, 0);
            __builtin_amdgcn_global_load_async_to_lds_b128((ASG i32x4*)(src + 8),
                                                           (ASL i32x4*)(dst + 8), 0, 0);
#else
            ((u32x4*)dst)[0] = *(const u32x4*)src;
            ((u32x4*)dst)[1] = *(const u32x4*)(src + 8);
#endif
            if (kbase + 2 * KSTEP <= K) __builtin_prefetch(src + KSTEP, 0, 1);
        } else {
            const u32x4 z = {0u, 0u, 0u, 0u};
            ((u32x4*)dst)[0] = z;
            ((u32x4*)dst)[1] = z;
        }
    }
    // ---- B: thread covers k0 = (tid/32)*4, n0 = (tid%32)*4; 4 coalesced b128 row loads,
    //         register 4x4 transpose, 4 packed b64 LDS stores (k-contiguous in [n][k] layout)
    {
        const int k0 = (tid >> 5) * 4;
        const int n0 = (tid & 31) * 4;
        const float* p = Bg + (size_t)(kbase + k0) * N + nbase + n0;
        const f32x4 r0 = *(const f32x4*)(p);
        const f32x4 r1 = *(const f32x4*)(p + N);
        const f32x4 r2 = *(const f32x4*)(p + 2 * N);
        const f32x4 r3 = *(const f32x4*)(p + 3 * N);
        if (kbase + 2 * KSTEP <= K) __builtin_prefetch(p + (size_t)KSTEP * N, 0, 1);
#pragma unroll
        for (int nn = 0; nn < 4; ++nn) {
            h4 o;
            o[0] = (_Float16)r0[nn];
            o[1] = (_Float16)r1[nn];
            o[2] = (_Float16)r2[nn];
            o[3] = (_Float16)r3[nn];
            *(h4*)(Bs + (n0 + nn) * LDA + k0) = o;
        }
    }
}

// D = act(A @ B + bias); A f16 [M,K] row-major (optionally gathered by rowIdx),
// B f32 [K,N] row-major (optionally per-expert-offset), out f16 or f32-scattered.
template <bool GATHER, bool EXPERT, bool ACT, bool OUTF16>
__global__ __launch_bounds__(256) void gemm_wmma_kernel(const _Float16* __restrict__ Ag,
                                                        const float* __restrict__ Bg,
                                                        const float* __restrict__ bias,
                                                        _Float16* __restrict__ outH,
                                                        float* __restrict__ outF,
                                                        const int* __restrict__ rowIdx,
                                                        const int* __restrict__ padOff,
                                                        int N, int K) {
    __shared__ alignas(16) _Float16 As[2][BM * LDA];
    __shared__ alignas(16) _Float16 Bs[2][BN * LDA];

    const int tid     = threadIdx.x;
    const int lane    = tid & 31;
    const int wave    = tid >> 5;
    const int waveM   = wave & 1;   // 2 row groups of 64
    const int waveN   = wave >> 1;  // 4 col groups of 32
    const int rowBase = blockIdx.y * BM;
    const int nbase   = blockIdx.x * BN;

    const float* Bptr  = Bg;
    const float* biasP = bias;
    if (EXPERT) {
        if (rowBase >= padOff[NEXP]) return;   // uniform early-exit (before any barrier)
        int e = 0;
#pragma unroll
        for (int kk = 1; kk <= NEXP; ++kk)
            e += (rowBase >= padOff[kk]) ? 1 : 0;
        Bptr  += (size_t)e * DIM * DIM;
        biasP += e * DIM;
    }

    int grow;  // global source row for this thread's A-staging slot
    {
        const int r = tid >> 1;
        grow = GATHER ? rowIdx[rowBase + r] : (rowBase + r);
    }

    const v8f zero = {0.f, 0.f, 0.f, 0.f, 0.f, 0.f, 0.f, 0.f};
    v8f acc[4][2];
#pragma unroll
    for (int i = 0; i < 4; ++i)
#pragma unroll
        for (int j = 0; j < 2; ++j) acc[i][j] = zero;

    const int NK = K / KSTEP;
    stage_tiles(Ag, Bptr, K, N, 0, grow, nbase, As[0], Bs[0], tid);
#if USE_ASYNC_LDS
    __builtin_amdgcn_s_wait_asynccnt(0);
#endif
    __syncthreads();

    for (int kt = 0; kt < NK; ++kt) {
        const int cur = kt & 1;
        v16h a[4], b[2];
#pragma unroll
        for (int i = 0; i < 4; ++i) a[i] = load_frag_a(As[cur], waveM * 4 + i, lane);
#pragma unroll
        for (int j = 0; j < 2; ++j) b[j] = load_frag_b(Bs[cur], waveN * 2 + j, lane);

        if (kt + 1 < NK)   // prefetch next K-slab into the other LDS buffer
            stage_tiles(Ag, Bptr, K, N, (kt + 1) * KSTEP, grow, nbase, As[cur ^ 1], Bs[cur ^ 1], tid);

#pragma unroll
        for (int i = 0; i < 4; ++i)
#pragma unroll
            for (int j = 0; j < 2; ++j)
                acc[i][j] = __builtin_amdgcn_wmma_f32_16x16x32_f16(
                    false, a[i], false, b[j], (short)0, acc[i][j], false, false);
#if USE_ASYNC_LDS
        __builtin_amdgcn_s_wait_asynccnt(0);
#endif
        __syncthreads();
    }

    // Epilogue: bias + (optional) GELU; C layout: VGPR r -> M = r + (lane/16)*8, N = lane%16
    const int colInTile = lane & 15;
    const int rhalf     = (lane >> 4) * 8;
#pragma unroll
    for (int j = 0; j < 2; ++j) {
        const int col  = nbase + (waveN * 2 + j) * 16 + colInTile;
        const float bv = biasP[col];
#pragma unroll
        for (int i = 0; i < 4; ++i) {
            const v8f c = acc[i][j];
#pragma unroll
            for (int r = 0; r < 8; ++r) {
                const int ml = waveM * 64 + i * 16 + rhalf + r;
                float v = c[r] + bv;
                if (ACT) v = gelu_exact(v);
                if (OUTF16) {
                    outH[(size_t)(rowBase + ml) * N + col] = (_Float16)v;
                } else {
                    const int g = rowIdx[rowBase + ml];
                    if (g >= 0) outF[(size_t)g * N + col] = v;
                }
            }
        }
    }
}

__global__ void cvt_kernel(const float* __restrict__ x, _Float16* __restrict__ xh, int n) {
    int i = blockIdx.x * blockDim.x + threadIdx.x;
    if (i < n) xh[i] = (_Float16)x[i];
}

__global__ void init_kernel(int* rowIdx, int* counts, int* fill) {
    int i = blockIdx.x * blockDim.x + threadIdx.x;
    if (i < MAXPAD) rowIdx[i] = -1;
    if (i < NEXP) { counts[i] = 0; fill[i] = 0; }
}

__global__ void count_kernel(const int* __restrict__ genre, int* counts) {
    int i = blockIdx.x * blockDim.x + threadIdx.x;
    if (i < BATCH) atomicAdd(&counts[genre[i]], 1);
}

__global__ void scan_kernel(const int* __restrict__ counts, int* padOff) {
    if (blockIdx.x == 0 && threadIdx.x == 0) {
        int off = 0;
        padOff[0] = 0;
        for (int k = 0; k < NEXP; ++k) {
            off += ((counts[k] + BM - 1) / BM) * BM;   // pad each segment to 128 rows
            padOff[k + 1] = off;
        }
    }
}

__global__ void scatter_kernel(const int* __restrict__ genre, const int* __restrict__ padOff,
                               int* fill, int* rowIdx) {
    int i = blockIdx.x * blockDim.x + threadIdx.x;
    if (i < BATCH) {
        int g = genre[i];
        int pos = padOff[g] + atomicAdd(&fill[g], 1);
        rowIdx[pos] = i;
    }
}

extern "C" void kernel_launch(void* const* d_in, const int* in_sizes, int n_in,
                              void* d_out, int out_size, void* d_ws, size_t ws_size,
                              hipStream_t stream) {
    (void)in_sizes; (void)n_in; (void)out_size; (void)ws_size;
    const float* x     = (const float*)d_in[0];
    const int*   genre = (const int*)d_in[1];
    const float* Ws1   = (const float*)d_in[2];
    const float* bs1   = (const float*)d_in[3];
    const float* Ws2   = (const float*)d_in[4];
    const float* bs2   = (const float*)d_in[5];
    const float* We1   = (const float*)d_in[6];
    const float* be1   = (const float*)d_in[7];
    const float* We2   = (const float*)d_in[8];
    const float* be2   = (const float*)d_in[9];
    float* out = (float*)d_out;

    char* ws = (char*)d_ws;
    size_t off = 0;
    auto alloc = [&](size_t bytes) -> void* {
        void* p = ws + off;
        off += (bytes + 255) & ~(size_t)255;
        return p;
    };
    _Float16* xh  = (_Float16*)alloc((size_t)BATCH * INDIM * 2);
    _Float16* h1h = (_Float16*)alloc((size_t)BATCH * DIM * 2);
    _Float16* sh  = (_Float16*)alloc((size_t)BATCH * DIM * 2);
    _Float16* th  = (_Float16*)alloc((size_t)MAXPAD * DIM * 2);
    int* rowIdx   = (int*)alloc(MAXPAD * 4);
    int* counts   = (int*)alloc(NEXP * 4);
    int* fill     = (int*)alloc(NEXP * 4);
    int* padOff   = (int*)alloc((NEXP + 1) * 4);

    // Expert dispatch bookkeeping + x conversion
    init_kernel<<<(MAXPAD + 255) / 256, 256, 0, stream>>>(rowIdx, counts, fill);
    cvt_kernel<<<(BATCH * INDIM + 255) / 256, 256, 0, stream>>>(x, xh, BATCH * INDIM);
    count_kernel<<<(BATCH + 255) / 256, 256, 0, stream>>>(genre, counts);
    scan_kernel<<<1, 32, 0, stream>>>(counts, padOff);
    scatter_kernel<<<(BATCH + 255) / 256, 256, 0, stream>>>(genre, padOff, fill, rowIdx);

    dim3 blk(256);
    dim3 gTrunk(DIM / BN, BATCH / BM);   // 8 x 32
    dim3 gExp(DIM / BN, MAXPAD / BM);    // 8 x 42 (blocks past padded total exit)

    // trunk: h1 = gelu(x @ Ws1 + bs1)
    gemm_wmma_kernel<false, false, true, true><<<gTrunk, blk, 0, stream>>>(
        xh, Ws1, bs1, h1h, nullptr, nullptr, nullptr, DIM, INDIM);
    // trunk: s = gelu(h1 @ Ws2 + bs2)
    gemm_wmma_kernel<false, false, true, true><<<gTrunk, blk, 0, stream>>>(
        h1h, Ws2, bs2, sh, nullptr, nullptr, nullptr, DIM, DIM);
    // experts: t = gelu(gather(s) @ We1[e] + be1[e])   (rows grouped/padded by expert)
    gemm_wmma_kernel<true, true, true, true><<<gExp, blk, 0, stream>>>(
        sh, We1, be1, th, nullptr, rowIdx, padOff, DIM, DIM);
    // experts: out[scatter] = t @ We2[e] + be2[e]
    gemm_wmma_kernel<false, true, false, false><<<gExp, blk, 0, stream>>>(
        th, We2, be2, nullptr, out, rowIdx, padOff, DIM, DIM);
}
